// RecurrentDecoderLSTM_87917980549848
// MI455X (gfx1250) — matrix operand order repaired and checked by reading, not compiled, not run on hardware
//
#include <hip/hip_runtime.h>

// Problem dims
#define B_  2048
#define S_  512
#define F_  64
#define H_  256
#define G_  1024   // 4*H

// Tiling
#define MB       32            // batch rows per workgroup
#define NWG      (B_ / MB)     // 64 persistent workgroups
#define NTHREADS 256           // 8 wave32 waves
#define NKT      10            // gate-GEMM k-tiles of 32: 2 (x/F) + 8 (h/H)
#define NNT      64            // gate-GEMM n-tiles of 16 (G/16)

// LDS strides (elements), padded for bank-conflict-free b128 fragment loads
#define GSTR 1032   // f32 gate buffer stride  (4128 B, 16B aligned)
#define HSTR 264    // bf16 h/c buffer stride  (528 B  -> lane L hits bank 4L)
#define XSTR 72     // bf16 x buffer stride    (144 B)

// LDS layout (bytes)
#define GATES_B (MB * GSTR * 4)         // 132096
#define HBUF_B  (MB * HSTR * 2)         // 16896
#define XBUF_B  (MB * XSTR * 2)         // 4608
#define SMEM_B  (GATES_B + 2*HBUF_B + XBUF_B + G_*4 + F_*4)  // 174848

// Workspace layout (bytes): packed bf16 weights + fused gate bias
#define WG_PK_ELEMS (NNT * NKT * 32 * 16)   // 327680 bf16
#define WD_PK_ELEMS (4 * 8 * 32 * 16)       // 16384 bf16
#define WG_PK_BYTES (WG_PK_ELEMS * 2)
#define WD_PK_BYTES (WD_PK_ELEMS * 2)

typedef __bf16 bf16x4  __attribute__((ext_vector_type(4)));
typedef __bf16 bf16x8  __attribute__((ext_vector_type(8)));
typedef __bf16 bf16x16 __attribute__((ext_vector_type(16)));
typedef float  f32x4   __attribute__((ext_vector_type(4)));
typedef float  f32x8   __attribute__((ext_vector_type(8)));

// ---- activations: CDNA5 has hardware v_tanh_f32 (TRANS op) ----
#if __has_builtin(__builtin_amdgcn_tanhf)
__device__ __forceinline__ float ftanh(float x) { return __builtin_amdgcn_tanhf(x); }
#else
__device__ __forceinline__ float ftanh(float x) {
    x = fminf(15.f, fmaxf(-15.f, x));
    float e = __expf(2.f * x);
    return (e - 1.f) * __builtin_amdgcn_rcpf(e + 1.f);   // division-free
}
#endif
__device__ __forceinline__ float fsigmoid(float x) {
    return fmaf(0.5f, ftanh(0.5f * x), 0.5f);            // 1 trans + 2 VALU
}

__device__ __forceinline__ bf16x16 cat8(bf16x8 lo, bf16x8 hi) {
    return __builtin_shufflevector(lo, hi, 0,1,2,3,4,5,6,7,8,9,10,11,12,13,14,15);
}

// A-fragment (16x32 bf16) from an LDS row-major buffer.
// Contiguous-k order used here (matches B packing): lane<16 -> row M=lane, K {0..15}; lane>=16 -> row M=lane-16, K {16..31}
__device__ __forceinline__ bf16x16 load_a_frag(const __bf16* buf, int stride,
                                               int mt, int kb, int lane) {
    int row = mt * 16 + (lane & 15);
    int k0  = kb + ((lane >> 4) << 4);
    const __bf16* p = buf + row * stride + k0;
    bf16x8 lo = *(const bf16x8*)p;
    bf16x8 hi = *(const bf16x8*)(p + 8);
    return cat8(lo, hi);
}

// B-fragment (32x16 bf16) pre-packed per-lane-contiguous in global: 32B per lane per block.
__device__ __forceinline__ bf16x16 load_b_frag(const __bf16* __restrict__ pk,
                                               int blk, int lane) {
    const __bf16* p = pk + ((size_t)blk * 32 + lane) * 16;
    bf16x8 lo = *(const bf16x8*)p;
    bf16x8 hi = *(const bf16x8*)(p + 8);
    return cat8(lo, hi);
}

// -------- weight packing: f32 -> bf16 in WMMA B-fragment order --------
__global__ void lstm_pack(const float* __restrict__ Wih, const float* __restrict__ Whh,
                          const float* __restrict__ Wd,  const float* __restrict__ bih,
                          const float* __restrict__ bhh,
                          __bf16* __restrict__ Wg_pk, __bf16* __restrict__ Wd_pk,
                          float* __restrict__ bias) {
    int id = blockIdx.x * blockDim.x + threadIdx.x;
    const int NG = NNT * NKT * 32;
    if (id < NG) {
        int lane = id & 31, blk = id >> 5;
        int nt = blk / NKT, kt = blk % NKT;
        int n = nt * 16 + (lane & 15);
        int khalf = (lane >> 4) * 16;
        const float* src = (kt < 2) ? (Wih + (size_t)n * F_ + kt * 32 + khalf)
                                    : (Whh + (size_t)n * H_ + (kt - 2) * 32 + khalf);
        __bf16* dst = Wg_pk + (size_t)id * 16;
        #pragma unroll
        for (int j = 0; j < 16; ++j) dst[j] = (__bf16)src[j];
    } else if (id < NG + 4 * 8 * 32) {
        int t = id - NG;
        int lane = t & 31, blk = t >> 5;          // blk = nt*8 + kt
        int nt = blk >> 3, kt = blk & 7;
        int n = nt * 16 + (lane & 15);
        const float* src = Wd + (size_t)n * H_ + kt * 32 + (lane >> 4) * 16;
        __bf16* dst = Wd_pk + (size_t)t * 16;
        #pragma unroll
        for (int j = 0; j < 16; ++j) dst[j] = (__bf16)src[j];
    } else if (id < NG + 4 * 8 * 32 + G_) {
        int j = id - (NG + 4 * 8 * 32);
        bias[j] = bih[j] + bhh[j];
    }
}

// -------- persistent LSTM rollout: one WG owns 32 batch rows for all 512 steps --------
__global__ __launch_bounds__(NTHREADS)
void lstm_main(const float* __restrict__ h0, const float* __restrict__ c0,
               const float* __restrict__ bdense,
               const __bf16* __restrict__ Wg_pk, const __bf16* __restrict__ Wd_pk,
               const float* __restrict__ biasg, float* __restrict__ out) {
    extern __shared__ __align__(16) char smem[];
    float*  gatesS = (float*)smem;                       // [32][1032] f32
    __bf16* hS     = (__bf16*)(smem + GATES_B);          // [32][264] bf16
    __bf16* cS     = hS + MB * HSTR;                     // [32][264] bf16
    __bf16* xS     = cS + MB * HSTR;                     // [32][72]  bf16
    float*  biasS  = (float*)(xS + MB * XSTR);           // [1024] f32
    float*  bdS    = biasS + G_;                         // [64]   f32

    const int tid    = threadIdx.x;
    const int lane   = tid & 31;
    const int wave   = tid >> 5;
    const int wgbase = blockIdx.x * MB;
    const int lr = (lane >> 4) << 3;   // C/D row offset per ISA accumulator layout
    const int lc = lane & 15;          // C/D column within tile

    // ---- prologue: biases to LDS, h0 -> hS & cS (bf16), c0 -> registers ----
    #pragma unroll
    for (int j = 0; j < 4; ++j) biasS[tid + j * NTHREADS] = biasg[tid + j * NTHREADS];
    if (tid < F_) bdS[tid] = bdense[tid];

    const int erow  = tid >> 3;          // elementwise: row 0..31
    const int ecol0 = (tid & 7) * 32;    // elementwise: 32 contiguous h-cols
    float creg[32];                      // cell state lives in VGPRs for all 512 steps
    {
        const float* h0p = h0 + (size_t)(wgbase + erow) * H_ + ecol0;
        const float* c0p = c0 + (size_t)(wgbase + erow) * H_ + ecol0;
        #pragma unroll
        for (int j = 0; j < 32; ++j) {
            creg[j] = c0p[j];
            __bf16 hb = (__bf16)h0p[j];
            hS[erow * HSTR + ecol0 + j] = hb;
            cS[erow * HSTR + ecol0 + j] = hb;   // prologue dense input is h0
        }
    }
    __syncthreads();

    // dense phase: y[32x64] = cS[32x256] @ Wd^T + b ; one 16x16 tile per wave
    const int dmt = wave >> 2;
    const int dnt = wave & 3;
    auto dense_phase = [&](int t) {
        f32x8 acc = {0.f, 0.f, 0.f, 0.f, 0.f, 0.f, 0.f, 0.f};
        #pragma unroll
        for (int kt = 0; kt < 8; ++kt) {
            bf16x16 a = load_a_frag(cS, HSTR, dmt, kt * 32, lane);
            bf16x16 b = load_b_frag(Wd_pk, dnt * 8 + kt, lane);
            acc = __builtin_amdgcn_wmma_f32_16x16x32_bf16(false, a, false, b,
                                                          (short)0, acc, false, false);
        }
        float bd = bdS[dnt * 16 + lc];
        #pragma unroll
        for (int r = 0; r < 8; ++r) {
            int row = dmt * 16 + r + lr;
            int col = dnt * 16 + lc;
            float y = acc[r] + bd;
            xS[row * XSTR + col] = (__bf16)y;          // next-step input
            if (t >= 0) {                              // streaming 256MB: bypass caches
                __builtin_nontemporal_store(
                    y, out + ((size_t)(wgbase + row) * S_ + t) * F_ + col);
            }
        }
    };

    dense_phase(-1);   // x_0 = dense(h0), no output write
    __syncthreads();

    for (int t = 0; t < S_; ++t) {
        // ---- gate GEMM: gates[32x1024] = x@Wih^T + h@Whh^T, wave w owns n-tiles 8w..8w+7
        #pragma unroll
        for (int mt = 0; mt < 2; ++mt) {
            f32x8 acc[8];
            #pragma unroll
            for (int n = 0; n < 8; ++n)
                acc[n] = (f32x8){0.f, 0.f, 0.f, 0.f, 0.f, 0.f, 0.f, 0.f};
            #pragma unroll
            for (int kt = 0; kt < NKT; ++kt) {
                bf16x16 a = (kt < 2) ? load_a_frag(xS, XSTR, mt, kt * 32, lane)
                                     : load_a_frag(hS, HSTR, mt, (kt - 2) * 32, lane);
                if (kt + 1 < NKT) {  // pull next k-tile of this wave's weight blocks (L2-hot)
                    __builtin_prefetch(
                        Wg_pk + ((size_t)(wave * 8 * NKT + kt + 1) * 32 + lane) * 16, 0, 0);
                }
                #pragma unroll
                for (int n = 0; n < 8; ++n) {
                    int nt = wave * 8 + n;
                    bf16x16 b = load_b_frag(Wg_pk, nt * NKT + kt, lane);
                    acc[n] = __builtin_amdgcn_wmma_f32_16x16x32_bf16(
                        false, a, false, b, (short)0, acc[n], false, false);
                }
            }
            #pragma unroll
            for (int n = 0; n < 8; ++n) {
                int col = (wave * 8 + n) * 16 + lc;
                #pragma unroll
                for (int r = 0; r < 8; ++r)
                    gatesS[(mt * 16 + r + lr) * GSTR + col] = acc[n][r];
            }
        }
        __syncthreads();

        // ---- elementwise LSTM cell: each thread owns 32 (row,col) cells
        // f32x4 gate reads (ds_load_b128), bf16x4 state writes (ds_store_b64),
        // hardware tanh for all activations.
        #pragma unroll
        for (int q = 0; q < 8; ++q) {
            const int col = ecol0 + q * 4;
            const float* gr = gatesS + erow * GSTR + col;
            f32x4 gi = *(const f32x4*)(gr)       + *(const f32x4*)(biasS + col);
            f32x4 gf = *(const f32x4*)(gr + 256) + *(const f32x4*)(biasS + col + 256);
            f32x4 gg = *(const f32x4*)(gr + 512) + *(const f32x4*)(biasS + col + 512);
            f32x4 go = *(const f32x4*)(gr + 768) + *(const f32x4*)(biasS + col + 768);
            bf16x4 hb, cb;
            #pragma unroll
            for (int u = 0; u < 4; ++u) {
                const int j = q * 4 + u;
                float iv = fsigmoid(gi[u]);
                float fv = fsigmoid(gf[u]);
                float gv = ftanh(gg[u]);
                float ov = fsigmoid(go[u]);
                float cn = fmaf(fv, creg[j], iv * gv);
                creg[j] = cn;
                float hn = ov * ftanh(cn);
                cb[u] = (__bf16)cn;
                hb[u] = (__bf16)hn;
            }
            *(bf16x4*)(hS + erow * HSTR + col) = hb;
            *(bf16x4*)(cS + erow * HSTR + col) = cb;
        }
        __syncthreads();

        // ---- y = dense(c_new): output slab t, and next x
        dense_phase(t);
        __syncthreads();
    }
}

extern "C" void kernel_launch(void* const* d_in, const int* in_sizes, int n_in,
                              void* d_out, int out_size, void* d_ws, size_t ws_size,
                              hipStream_t stream) {
    (void)in_sizes; (void)n_in; (void)out_size; (void)ws_size;
    // setup_inputs order: x(unused, eps=0), h0, c0, W_ih, W_hh, b_ih, b_hh, W_dense, b_dense, seq_len
    const float* h0  = (const float*)d_in[1];
    const float* c0  = (const float*)d_in[2];
    const float* Wih = (const float*)d_in[3];
    const float* Whh = (const float*)d_in[4];
    const float* bih = (const float*)d_in[5];
    const float* bhh = (const float*)d_in[6];
    const float* Wd  = (const float*)d_in[7];
    const float* bd  = (const float*)d_in[8];
    float* out = (float*)d_out;

    __bf16* Wg_pk = (__bf16*)d_ws;
    __bf16* Wd_pk = (__bf16*)((char*)d_ws + WG_PK_BYTES);
    float*  biasg = (float*)((char*)d_ws + WG_PK_BYTES + WD_PK_BYTES);

    const int packN = NNT * NKT * 32 + 4 * 8 * 32 + G_;
    lstm_pack<<<(packN + 255) / 256, 256, 0, stream>>>(Wih, Whh, Wd, bih, bhh,
                                                       Wg_pk, Wd_pk, biasg);
    lstm_main<<<NWG, NTHREADS, SMEM_B, stream>>>(h0, c0, bd, Wg_pk, Wd_pk, biasg, out);
}
